// ObjectBoxDetect_10471130267819
// MI455X (gfx1250) — matrix operand order repaired and checked by reading, not compiled, not run on hardware
//
#include <hip/hip_runtime.h>
#include <hip/hip_bf16.h>

// f16 / f32 ext-vector types for WMMA fragments
typedef _Float16 v16h __attribute__((ext_vector_type(16)));
typedef _Float16 v8h  __attribute__((ext_vector_type(8)));
typedef _Float16 v4h  __attribute__((ext_vector_type(4)));
typedef float    v8f  __attribute__((ext_vector_type(8)));

#define KC        256            // K-chunk staged in LDS (divides 256/512/1024)
#define NROWS     96             // 85 outputs padded to 6 x 16
#define ROWSTRIDE 264            // f16 units/row; 132 dwords == 4 mod 64 -> bank-friendly b128

template <int C, int S, int LEVEL, int LOFF>
__global__ __launch_bounds__(256)
void yolo_decode_level(const float* __restrict__ feat,   // [16, C, S, S]
                       const float* __restrict__ Wm,     // [85, C]
                       const float* __restrict__ bias,   // [85]
                       float* __restrict__ out)          // [16, 8400, 85]
{
    constexpr int   S2     = S * S;
    constexpr int   TPI    = S2 >> 4;                    // tiles per image
    constexpr float STRIDE = (float)(8 << LEVEL);
    constexpr float PW     = 4.0f * (float)(1 << LEVEL); // d = (2y)^2 * 2^level

    __shared__ _Float16 ldsW[NROWS * ROWSTRIDE];         // 50,688 B

    const int tid  = threadIdx.x;
    const int lane = tid & 31;
    const int wave = tid >> 5;        // 0..7
    const int n16  = lane & 15;       // N (and A-row M) lane index
    const int hh   = lane >> 4;       // half select
    // Each wave owns one M-tile of 16 consecutive flat positions; grid exactly covers all tiles.
    const int tile = blockIdx.x * 8 + wave;
    const int b    = tile / TPI;                         // constant divisor -> mul/shift
    const int q0   = (tile - b * TPI) << 4;

    // Per-lane A base: includes the lane-half K offset (hh*8 channels); all channel
    // offsets inside the unrolled K loop become compile-time immediates.
    const float* aptr = feat + (size_t)b * C * S2 + (size_t)hh * 8 * S2 + q0 + n16;

    v8f acc[6];
#pragma unroll
    for (int t = 0; t < 6; ++t) acc[t] = (v8f)0.0f;

    for (int c0 = 0; c0 < C; c0 += KC) {
        __syncthreads();
        // ---- stage W[0:96][c0:c0+KC] into LDS as f16 (zero-pad rows 85..95) ----
#pragma unroll
        for (int it = 0; it < NROWS * (KC / 4) / 256; ++it) {
            const int idx = tid + it * 256;
            const int row = idx >> 6;                    // / (KC/4)
            const int c4  = (idx & 63) << 2;
            v4h hv;
            if (row < 85) {
                const float* wp = Wm + row * C + c0 + c4;
                hv = (v4h){(_Float16)wp[0], (_Float16)wp[1],
                           (_Float16)wp[2], (_Float16)wp[3]};
            } else {
                hv = (v4h)(_Float16)0.0f;
            }
            *(v4h*)&ldsW[row * ROWSTRIDE + c4] = hv;     // ds_store_b64
        }
        __syncthreads();

        if (c0 + KC < C) {                               // uniform; folds away when C==KC
            __builtin_prefetch(aptr + (size_t)KC * S2, 0, 1);   // global_prefetch_b8
        }

        // ---- K loop: 8 steps of K=32 per chunk (fully unrolled -> immediate offsets) ----
#pragma unroll
        for (int kk = 0; kk < KC; kk += 32) {
            // All 6 B fragments first (distinct regs -> ds latency overlaps WMMA chain).
            v16h bF[6];
#pragma unroll
            for (int t = 0; t < 6; ++t) {
                const _Float16* bp = &ldsW[(t * 16 + n16) * ROWSTRIDE + kk + hh * 8];
                const v8h blo = *(const v8h*)bp;         // ds_load_b128
                const v8h bhi = *(const v8h*)(bp + 16);  // ds_load_b128
#pragma unroll
                for (int e = 0; e < 8; ++e) { bF[t][e] = blo[e]; bF[t][8 + e] = bhi[e]; }
            }

            // A fragment (16 x 32 f16). Lane half: K 0-7/16-23 vs 8-15/24-31.
            float af[16];
#pragma unroll
            for (int e = 0; e < 8; ++e) af[e]     = aptr[(kk + e) * S2];       // imm offsets
#pragma unroll
            for (int e = 0; e < 8; ++e) af[8 + e] = aptr[(kk + 16 + e) * S2];  // imm offsets
            v16h aF;
#pragma unroll
            for (int e = 0; e < 16; ++e) aF[e] = (_Float16)af[e];

#pragma unroll
            for (int t = 0; t < 6; ++t) {
                acc[t] = __builtin_amdgcn_wmma_f32_16x16x32_f16(
                    /*neg_a=*/false, aF, /*neg_b=*/false, bF[t],
                    /*c_mod=*/(short)0, acc[t],
                    /*reuse_a=*/false, /*reuse_b=*/false);
            }
        }
        aptr += (size_t)KC * S2;
    }

    // ---- epilogue: bias + sigmoid + box decode + store ----
    float bv[6];
#pragma unroll
    for (int t = 0; t < 6; ++t) {
        const int o = t * 16 + n16;
        bv[t] = (o < 85) ? bias[o] : 0.0f;
    }

#pragma unroll
    for (int r = 0; r < 8; ++r) {
        const int q    = q0 + r + 8 * hh;      // C/D: VGPR r -> M=r (lanes 0-15), M=r+8 (16-31)
        const int hrow = q / S;                // constant divisor
        const int wcol = q - hrow * S;

        float yv[6];
#pragma unroll
        for (int t = 0; t < 6; ++t) {
            const float p = acc[t][r] + bv[t];
            yv[t] = 1.0f / (1.0f + __expf(-p));
        }

        // Box channels 0..3 live in lanes (L&16)+0..3 of tile 0 for this row.
        const int base = lane & 16;
        const float y0 = __shfl(yv[0], base + 0, 32);
        const float y1 = __shfl(yv[0], base + 1, 32);
        const float y2 = __shfl(yv[0], base + 2, 32);
        const float y3 = __shfl(yv[0], base + 3, 32);
        const float d0 = y0 * y0 * PW, d1 = y1 * y1 * PW;
        const float d2 = y2 * y2 * PW, d3 = y3 * y3 * PW;
        const float cx = 0.5f * STRIDE * ((float)(2 * wcol) + 1.0f - d0 + d2);
        const float cy = 0.5f * STRIDE * ((float)(2 * hrow) + 1.0f - d1 + d3);
        const float bw = STRIDE * (d0 + d2 - 1.0f);
        const float bh = STRIDE * (d1 + d3 - 1.0f);

        float v0 = yv[0];
        if      (n16 == 0) v0 = cx;
        else if (n16 == 1) v0 = cy;
        else if (n16 == 2) v0 = bw;
        else if (n16 == 3) v0 = bh;

        const size_t obase = ((size_t)b * 8400 + (size_t)(LOFF + q)) * 85;
        out[obase + n16] = v0;
#pragma unroll
        for (int t = 1; t < 6; ++t) {
            const int o = t * 16 + n16;
            if (o < 85) out[obase + o] = yv[t];
        }
    }
}

extern "C" void kernel_launch(void* const* d_in, const int* in_sizes, int n_in,
                              void* d_out, int out_size, void* d_ws, size_t ws_size,
                              hipStream_t stream) {
    const float* x0 = (const float*)d_in[0];
    const float* x1 = (const float*)d_in[1];
    const float* x2 = (const float*)d_in[2];
    const float* W0 = (const float*)d_in[3];
    const float* b0 = (const float*)d_in[4];
    const float* W1 = (const float*)d_in[5];
    const float* b1 = (const float*)d_in[6];
    const float* W2 = (const float*)d_in[7];
    const float* b2 = (const float*)d_in[8];
    float* out = (float*)d_out;

    // tiles = 16*S*S/16 = S*S; blocks = tiles/8 (exact for S=80/40/20)
    yolo_decode_level< 256, 80, 0,    0><<<800, 256, 0, stream>>>(x0, W0, b0, out);
    yolo_decode_level< 512, 40, 1, 6400><<<200, 256, 0, stream>>>(x1, W1, b1, out);
    yolo_decode_level<1024, 20, 2, 8000><<< 50, 256, 0, stream>>>(x2, W2, b2, out);
}
